// GraphAttentionEmbedding_32332513804699
// MI455X (gfx1250) — matrix operand order repaired and checked by reading, not compiled, not run on hardware
//
#include <hip/hip_runtime.h>
#include <hip/hip_bf16.h>

// ---------- problem constants ----------
#define NN   100000
#define EE   600000
#define INC  128
#define OUTC 128
#define HH   2
#define CC   64
#define MSGD 64
#define TD   64
#define EDD  128

typedef __attribute__((ext_vector_type(16))) _Float16 v16h_t;
typedef __attribute__((ext_vector_type(8)))  _Float16 v8h_t;
typedef __attribute__((ext_vector_type(8)))  float    v8f_t;

// ---------- workspace layout (bytes) ----------
#define WS_XH    ((size_t)0)                         // N*128 f16      = 25,600,000
#define WS_WN    (WS_XH  + (size_t)NN*128*2)         // 65536 f16      =    131,072
#define WS_WE    (WS_WN  + (size_t)65536*2)          // 16384 f16      =     32,768
#define WS_Q     (WS_WE  + (size_t)16384*2)          // N*128 f32
#define WS_K     (WS_Q   + (size_t)NN*128*4)
#define WS_V     (WS_K   + (size_t)NN*128*4)
#define WS_EX    (WS_V   + (size_t)NN*128*4)         // E*2 f32
#define WS_DEN   (WS_EX  + (size_t)EE*2*4)           // N*2 f32

// =====================================================================
// 1) x (f32) -> xh (f16), row-major [N,128]
// =====================================================================
__global__ __launch_bounds__(256) void cvt_x_kernel(const float* __restrict__ x,
                                                    _Float16* __restrict__ xh) {
    size_t base = ((size_t)blockIdx.x * 256 + threadIdx.x) * 8;
    #pragma unroll
    for (int i = 0; i < 8; ++i) xh[base + i] = (_Float16)x[base + i];
}

// =====================================================================
// 2) pack node weights [Wq|Wk|Wv|Wskip] (each 128x128 row-major, row=k)
//    into WMMA B-fragment order: frag f = ntile*4+kc, 512 halves per frag,
//    half (lane,i): col n = ntile*16+(lane&15); k = kc*32 + (lane<16 ? 0 : 8) + (i<8 ? i : 8+i)
// =====================================================================
__global__ __launch_bounds__(256) void pack_wnode_kernel(const float* __restrict__ Wq,
                                                         const float* __restrict__ Wk,
                                                         const float* __restrict__ Wv,
                                                         const float* __restrict__ Ws,
                                                         _Float16* __restrict__ Wp) {
    int idx = blockIdx.x * 256 + threadIdx.x;          // 0..65535
    int f      = idx >> 9;
    int within = idx & 511;
    int vlane  = within >> 4;
    int i      = within & 15;
    int ntile  = f >> 2;                               // 0..31 (512 cols total)
    int kc     = f & 3;
    int n      = ntile * 16 + (vlane & 15);            // 0..511 combined col
    int kk     = kc * 32 + ((vlane < 16) ? 0 : 8) + ((i < 8) ? i : 8 + i);
    int grp    = n >> 7;
    int ncol   = n & 127;
    const float* W = (grp == 0) ? Wq : (grp == 1) ? Wk : (grp == 2) ? Wv : Ws;
    Wp[idx] = (_Float16)W[(size_t)kk * 128 + ncol];
}

// same packing for We (128x128), 8 ntiles -> 16384 halves
__global__ __launch_bounds__(256) void pack_wedge_kernel(const float* __restrict__ We,
                                                         _Float16* __restrict__ Wp) {
    int idx = blockIdx.x * 256 + threadIdx.x;          // 0..16383
    int f      = idx >> 9;
    int within = idx & 511;
    int vlane  = within >> 4;
    int i      = within & 15;
    int ntile  = f >> 2;                               // 0..7
    int kc     = f & 3;
    int n      = ntile * 16 + (vlane & 15);
    int kk     = kc * 32 + ((vlane < 16) ? 0 : 8) + ((i < 8) ? i : 8 + i);
    Wp[idx] = (_Float16)We[(size_t)kk * 128 + n];
}

// =====================================================================
// 3) zero softmax denominators
// =====================================================================
__global__ __launch_bounds__(256) void zero_denom_kernel(float* __restrict__ den) {
    int idx = blockIdx.x * 256 + threadIdx.x;
    if (idx < NN * HH) den[idx] = 0.0f;
}

// =====================================================================
// 4) node projections: [N,128] x [128,512] with v_wmma_f32_16x16x32_f16.
//    One wave per 16x16 output tile, K=128 -> 4 WMMAs. Skip proj -> d_out.
// =====================================================================
__global__ __launch_bounds__(256) void node_proj_kernel(const _Float16* __restrict__ xh,
                                                        const _Float16* __restrict__ Wp,
                                                        const float* __restrict__ bq,
                                                        const float* __restrict__ bk,
                                                        const float* __restrict__ bv,
                                                        const float* __restrict__ bs,
                                                        float* __restrict__ q,
                                                        float* __restrict__ k,
                                                        float* __restrict__ v,
                                                        float* __restrict__ outskip) {
    const int lane  = threadIdx.x & 31;
    const int gid   = blockIdx.x * 8 + (threadIdx.x >> 5);  // 200000 waves exact
    const int mtile = gid >> 5;                              // 0..6249
    const int ntile = gid & 31;                              // 0..31
    const int row   = mtile * 16 + (lane & 15);
    const int abase = (lane & 16) ? 8 : 0;

    v8f_t c = {};
    #pragma unroll
    for (int kc = 0; kc < 4; ++kc) {
        const _Float16* ap = xh + (size_t)row * 128 + kc * 32 + abase;
        v8h_t alo = *(const v8h_t*)(ap);
        v8h_t ahi = *(const v8h_t*)(ap + 16);
        v16h_t a = __builtin_shufflevector(alo, ahi, 0,1,2,3,4,5,6,7,8,9,10,11,12,13,14,15);
        v16h_t b = *(const v16h_t*)(Wp + ((size_t)(ntile * 4 + kc) * 512 + lane * 16));
        c = __builtin_amdgcn_wmma_f32_16x16x32_f16(false, a, false, b, (short)0, c, false, false);
    }

    const int grp  = ntile >> 3;                 // 0=q 1=k 2=v 3=skip
    const int ncol = (ntile & 7) * 16 + (lane & 15);
    const float* bsel = (grp == 0) ? bq : (grp == 1) ? bk : (grp == 2) ? bv : bs;
    float* dsel       = (grp == 0) ? q  : (grp == 1) ? k  : (grp == 2) ? v  : outskip;
    const float bias  = bsel[ncol];
    const int mofs    = (lane & 16) ? 8 : 0;
    #pragma unroll
    for (int r = 0; r < 8; ++r)
        dsel[(size_t)(mtile * 16 + mofs + r) * 128 + ncol] = c[r] + bias;
}

// =====================================================================
// edge A-fragments: edge_attr row = [cos(rel_t*wt+bt) (64) | msg (64)] -> f16
// lane (l&15) owns edge row M; K-halves follow the 16-bit A layout.
// =====================================================================
__device__ inline void build_edge_afrags(int tile, int lane,
                                         const int* __restrict__ ei,
                                         const float* __restrict__ lastup,
                                         const float* __restrict__ tarr,
                                         const float* __restrict__ msg,
                                         const float* __restrict__ wt,
                                         const float* __restrict__ bt,
                                         v16h_t a[4]) {
    const int e0   = tile * 16 + (lane & 15);
    const int s0   = ei[e0];
    const float rel = lastup[s0] - tarr[e0];
    const int base = (lane & 16) ? 8 : 0;
    #pragma unroll
    for (int kc = 0; kc < 2; ++kc) {                  // time-encoding half
        v16h_t av;
        #pragma unroll
        for (int i = 0; i < 16; ++i) {
            int kk = kc * 32 + base + ((i < 8) ? i : 8 + i);
            av[i] = (_Float16)cosf(rel * wt[kk] + bt[kk]);
        }
        a[kc] = av;
    }
    const float* mrow = msg + (size_t)e0 * MSGD;
    #pragma unroll
    for (int kc = 2; kc < 4; ++kc) {                  // msg half
        v16h_t av;
        #pragma unroll
        for (int i = 0; i < 16; ++i) {
            int kk = kc * 32 + base + ((i < 8) ? i : 8 + i) - 64;
            av[i] = (_Float16)mrow[kk];
        }
        a[kc] = av;
    }
}

// =====================================================================
// 5) edge pass 1: e = edge_attr@We (WMMA), alpha = q[dst]·(k[src]+e)/8,
//    ex = exp(alpha) -> exbuf; denom[dst,h] += ex (f32 atomics, L2-resident)
// =====================================================================
__global__ __launch_bounds__(256) void edge_alpha_kernel(const int* __restrict__ ei,
                                                         const float* __restrict__ lastup,
                                                         const float* __restrict__ tarr,
                                                         const float* __restrict__ msg,
                                                         const float* __restrict__ wt,
                                                         const float* __restrict__ bt,
                                                         const _Float16* __restrict__ Wep,
                                                         const float* __restrict__ q,
                                                         const float* __restrict__ k,
                                                         float* __restrict__ exbuf,
                                                         float* __restrict__ denom) {
    const int lane = threadIdx.x & 31;
    const int gid  = blockIdx.x * 8 + (threadIdx.x >> 5);
    if (gid >= EE / 16) return;                       // wave-uniform guard
    const int tile = gid;

    v16h_t a[4];
    build_edge_afrags(tile, lane, ei, lastup, tarr, msg, wt, bt, a);

    const int mofs = (lane & 16) ? 8 : 0;
    int rsrc[8], rdst[8];
    #pragma unroll
    for (int r = 0; r < 8; ++r) {
        int eidx = tile * 16 + mofs + r;
        rsrc[r] = ei[eidx];
        rdst[r] = ei[EE + eidx];
    }
    const int col16 = lane & 15;

    #pragma unroll
    for (int h = 0; h < 2; ++h) {
        float pacc[8];
        #pragma unroll
        for (int r = 0; r < 8; ++r) pacc[r] = 0.0f;
        #pragma unroll
        for (int nt = 0; nt < 4; ++nt) {
            const int ntile = h * 4 + nt;
            v8f_t c = {};
            #pragma unroll
            for (int kc = 0; kc < 4; ++kc) {
                v16h_t b = *(const v16h_t*)(Wep + ((size_t)(ntile * 4 + kc) * 512 + lane * 16));
                c = __builtin_amdgcn_wmma_f32_16x16x32_f16(false, a[kc], false, b, (short)0, c, false, false);
            }
            const int col = ntile * 16 + col16;
            #pragma unroll
            for (int r = 0; r < 8; ++r) {
                float kj = k[(size_t)rsrc[r] * 128 + col] + c[r];
                float qi = q[(size_t)rdst[r] * 128 + col];
                pacc[r] += qi * kj;
            }
        }
        #pragma unroll
        for (int r = 0; r < 8; ++r) {                 // sum over 16 columns
            float s = pacc[r];
            s += __shfl_xor(s, 1, 32);
            s += __shfl_xor(s, 2, 32);
            s += __shfl_xor(s, 4, 32);
            s += __shfl_xor(s, 8, 32);
            if (col16 == 0) {
                int eidx = tile * 16 + mofs + r;
                float exv = expf(s * 0.125f);         // 1/sqrt(64)
                exbuf[(size_t)eidx * 2 + h] = exv;
                atomicAdd(&denom[(size_t)rdst[r] * 2 + h], exv);
            }
        }
    }
}

// =====================================================================
// 6) edge pass 2: recompute e (WMMA), attn = ex/denom[dst],
//    out[dst] += attn * (v[src] + e)   (f32 global atomics)
// =====================================================================
__global__ __launch_bounds__(256) void edge_aggr_kernel(const int* __restrict__ ei,
                                                        const float* __restrict__ lastup,
                                                        const float* __restrict__ tarr,
                                                        const float* __restrict__ msg,
                                                        const float* __restrict__ wt,
                                                        const float* __restrict__ bt,
                                                        const _Float16* __restrict__ Wep,
                                                        const float* __restrict__ v,
                                                        const float* __restrict__ exbuf,
                                                        const float* __restrict__ denom,
                                                        float* __restrict__ out) {
    const int lane = threadIdx.x & 31;
    const int gid  = blockIdx.x * 8 + (threadIdx.x >> 5);
    if (gid >= EE / 16) return;
    const int tile = gid;

    v16h_t a[4];
    build_edge_afrags(tile, lane, ei, lastup, tarr, msg, wt, bt, a);

    const int mofs = (lane & 16) ? 8 : 0;
    int rsrc[8], rdst[8];
    float attn[2][8];
    #pragma unroll
    for (int r = 0; r < 8; ++r) {
        int eidx = tile * 16 + mofs + r;
        rsrc[r] = ei[eidx];
        rdst[r] = ei[EE + eidx];
        #pragma unroll
        for (int h = 0; h < 2; ++h)
            attn[h][r] = exbuf[(size_t)eidx * 2 + h] /
                         (denom[(size_t)rdst[r] * 2 + h] + 1e-16f);
    }
    const int col16 = lane & 15;

    #pragma unroll
    for (int ntile = 0; ntile < 8; ++ntile) {
        const int h = ntile >> 2;
        v8f_t c = {};
        #pragma unroll
        for (int kc = 0; kc < 4; ++kc) {
            v16h_t b = *(const v16h_t*)(Wep + ((size_t)(ntile * 4 + kc) * 512 + lane * 16));
            c = __builtin_amdgcn_wmma_f32_16x16x32_f16(false, a[kc], false, b, (short)0, c, false, false);
        }
        const int col = ntile * 16 + col16;
        #pragma unroll
        for (int r = 0; r < 8; ++r) {
            float vj  = v[(size_t)rsrc[r] * 128 + col] + c[r];
            float val = attn[h][r] * vj;
            atomicAdd(&out[(size_t)rdst[r] * 128 + col], val);
        }
    }
}

// =====================================================================
extern "C" void kernel_launch(void* const* d_in, const int* in_sizes, int n_in,
                              void* d_out, int out_size, void* d_ws, size_t ws_size,
                              hipStream_t stream) {
    const float* x       = (const float*)d_in[0];
    const float* lastup  = (const float*)d_in[1];
    const int*   ei      = (const int*)  d_in[2];
    const float* tarr    = (const float*)d_in[3];
    const float* msg     = (const float*)d_in[4];
    const float* wt      = (const float*)d_in[5];
    const float* bt      = (const float*)d_in[6];
    const float* Wq      = (const float*)d_in[7];
    const float* bq      = (const float*)d_in[8];
    const float* Wk      = (const float*)d_in[9];
    const float* bk      = (const float*)d_in[10];
    const float* Wv      = (const float*)d_in[11];
    const float* bv      = (const float*)d_in[12];
    const float* We      = (const float*)d_in[13];
    const float* Ws      = (const float*)d_in[14];
    const float* bs      = (const float*)d_in[15];
    float* out = (float*)d_out;

    char* ws = (char*)d_ws;
    _Float16* xh  = (_Float16*)(ws + WS_XH);
    _Float16* Wn  = (_Float16*)(ws + WS_WN);
    _Float16* Wep = (_Float16*)(ws + WS_WE);
    float* q      = (float*)(ws + WS_Q);
    float* k      = (float*)(ws + WS_K);
    float* v      = (float*)(ws + WS_V);
    float* exbuf  = (float*)(ws + WS_EX);
    float* denom  = (float*)(ws + WS_DEN);

    // preprocessing
    cvt_x_kernel<<<6250, 256, 0, stream>>>(x, xh);                  // 6250*256*8 = N*128
    pack_wnode_kernel<<<256, 256, 0, stream>>>(Wq, Wk, Wv, Ws, Wn); // 65536
    pack_wedge_kernel<<<64, 256, 0, stream>>>(We, Wep);             // 16384
    zero_denom_kernel<<<(NN * HH + 255) / 256, 256, 0, stream>>>(denom);

    // node projections (skip proj initializes d_out)
    node_proj_kernel<<<25000, 256, 0, stream>>>(xh, Wn, bq, bk, bv, bs, q, k, v, out);

    // edge passes
    const int etiles  = EE / 16;                 // 37500
    const int eblocks = (etiles + 7) / 8;        // 4688
    edge_alpha_kernel<<<eblocks, 256, 0, stream>>>(ei, lastup, tarr, msg, wt, bt,
                                                   Wep, q, k, exbuf, denom);
    edge_aggr_kernel<<<eblocks, 256, 0, stream>>>(ei, lastup, tarr, msg, wt, bt,
                                                  Wep, v, exbuf, denom, out);
}
